// DenseShift_55396488183982
// MI455X (gfx1250) — matrix operand order repaired and checked by reading, not compiled, not run on hardware
//
#include <hip/hip_runtime.h>

typedef __attribute__((ext_vector_type(16))) _Float16 v16h;
typedef __attribute__((ext_vector_type(8)))  float    v8f;
typedef __attribute__((ext_vector_type(4)))  float    v4f;

#define IN_DIM  256
#define OUT_DIM 256
#define KT 8               // 256 / 32 K-slices total
#define KT_PHASE 4         // K-slices staged per LDS phase (64 KB)
#define NT 16              // 256 / 16 N tiles
#define WAVES 8            // waves per block
#define ROWS_PER_BLOCK (WAVES * 16)   // 128

// LDS: one phase of packed W = 4 kt * 16 nt * 32 lanes * 16 halfs = 64 KB
#define LDS_HALFS (KT_PHASE * NT * 32 * 16)

// ---------------------------------------------------------------------------
// Pre-pack W = (-1)^sign * 2^shift into f16, laid out EXACTLY as the per-lane
// B-matrix fragments this code feeds V_WMMA_F32_16X16X32_F16 (contiguous k, same order as A):
//   fragment (kt, nt), lane L holds 16 halfs b[i] with
//     N = nt*16 + (L & 15),  K = kt*32 + (L >> 4)*16 + i
// Flat layout: wp[ ((kt*NT + nt)*32 + lane)*16 + i ]   (131072 B in d_ws)
// ---------------------------------------------------------------------------
__global__ void pack_w_kernel(const float* __restrict__ shift,
                              const float* __restrict__ sign,
                              _Float16* __restrict__ wp) {
    int tid = blockIdx.x * blockDim.x + threadIdx.x;   // [0, KT*NT*32)
    if (tid >= KT * NT * 32) return;
    int lane = tid & 31;
    int nt   = (tid >> 5) & 15;
    int kt   = tid >> 9;
    int n     = nt * 16 + (lane & 15);
    int kbase = kt * 32 + (lane >> 4) * 16;
    _Float16* dst = wp + (size_t)tid * 16;
#pragma unroll
    for (int i = 0; i < 16; ++i) {
        int k = kbase + i;
        float sv = shift[k * OUT_DIM + n];   // integer in [-10, -1]
        float gv = sign [k * OUT_DIM + n];   // 0.0 or 1.0
        float w = ldexpf(1.0f, (int)sv);
        if (gv > 0.5f) w = -w;
        dst[i] = (_Float16)w;                // exact: +-2^k
    }
}

// ---------------------------------------------------------------------------
// Main GEMM: each wave computes a 16(M) x 256(N) output strip.
// W staged global->LDS in two 64KB phases (DScnt path, decoupled from the
// streaming A loads on LOADcnt). B fragments double-buffered from LDS and A
// fragments software-pipelined across kt so waits can be partial.
// 8 K-slices x 16 N-tiles = 128 v_wmma_f32_16x16x32_f16 per wave.
// ---------------------------------------------------------------------------
__global__ __launch_bounds__(WAVES * 32)
void denseshift_gemm(const float* __restrict__ x,
                     const _Float16* __restrict__ wp,
                     const float* __restrict__ bias,
                     float* __restrict__ out) {
    __shared__ alignas(16) _Float16 ldsW[LDS_HALFS];   // 64 KB

    const int lane  = threadIdx.x & 31;
    const int wave  = threadIdx.x >> 5;
    const size_t m0 = ((size_t)blockIdx.x * WAVES + wave) * 16;

    const int row   = lane & 15;        // A: M row this lane holds; B/C: N col
    const int khalf = (lane >> 4) * 16; // A-fragment K sub-chunk select
    const float* xrow = x + (m0 + row) * IN_DIM;

    v8f acc[NT] = {};                   // 16 tiles x 8 VGPRs f32 accumulators

    for (int p = 0; p < 2; ++p) {
        // ---- stage one 64KB phase of packed W into LDS ----
        if (p) __syncthreads();         // all waves done reading previous phase
        {
            const v4f* src = (const v4f*)wp + (size_t)p * (LDS_HALFS / 8);
            v4f* dst = (v4f*)ldsW;
#pragma unroll
            for (int i = 0; i < LDS_HALFS / 8 / (WAVES * 32); ++i) {
                int idx = threadIdx.x + i * (WAVES * 32);
                dst[idx] = src[idx];
            }
        }
        __syncthreads();

        // ---- preload A raw data for first kt of this phase ----
        const float* pa0 = xrow + (p * KT_PHASE) * 32 + khalf;
        v4f r0 = __builtin_nontemporal_load((const v4f*)(pa0));
        v4f r1 = __builtin_nontemporal_load((const v4f*)(pa0 + 4));
        v4f r2 = __builtin_nontemporal_load((const v4f*)(pa0 + 8));
        v4f r3 = __builtin_nontemporal_load((const v4f*)(pa0 + 12));

#pragma unroll
        for (int kt4 = 0; kt4 < KT_PHASE; ++kt4) {
            // ---- prefetch next kt's A raw data (distinct regs) ----
            v4f n0, n1, n2, n3;
            if (kt4 + 1 < KT_PHASE) {
                const float* pa = xrow + (p * KT_PHASE + kt4 + 1) * 32 + khalf;
                n0 = __builtin_nontemporal_load((const v4f*)(pa));
                n1 = __builtin_nontemporal_load((const v4f*)(pa + 4));
                n2 = __builtin_nontemporal_load((const v4f*)(pa + 8));
                n3 = __builtin_nontemporal_load((const v4f*)(pa + 12));
            } else {
                n0 = r0; n1 = r1; n2 = r2; n3 = r3;
            }
            // ---- convert current A to f16 fragment (contiguous k = 16h + s, as packed B) ----
            v16h a;
#pragma unroll
            for (int i = 0; i < 4; ++i) {
                a[i]      = (_Float16)r0[i];
                a[4 + i]  = (_Float16)r1[i];
                a[8 + i]  = (_Float16)r2[i];
                a[12 + i] = (_Float16)r3[i];
            }
            // ---- 16 N-tiles reuse this A fragment; B double-buffered ----
            const _Float16* bl = ldsW + ((size_t)(kt4 * NT) * 32 + lane) * 16;
            v16h b = *(const v16h*)(bl);
#pragma unroll
            for (int nt = 0; nt < NT; ++nt) {
                v16h bn;
                if (nt + 1 < NT) bn = *(const v16h*)(bl + (size_t)(nt + 1) * 32 * 16);
                else             bn = b;
                acc[nt] = __builtin_amdgcn_wmma_f32_16x16x32_f16(
                    false, a, false, b, (short)0, acc[nt], false, false);
                b = bn;
            }
            r0 = n0; r1 = n1; r2 = n2; r3 = n3;
        }
    }

    // ---- Epilogue: bias add + store. C layout: VGPR r -> M = r + 8*(lane>=16),
    //      N = lane&15 (per ISA C/D table). Streamed NT stores. ----
    const int mhi = (lane >> 4) * 8;
#pragma unroll
    for (int nt = 0; nt < NT; ++nt) {
        const int n  = nt * 16 + row;
        const float bv = bias[n];
#pragma unroll
        for (int r = 0; r < 8; ++r) {
            __builtin_nontemporal_store(acc[nt][r] + bv,
                                        &out[(m0 + mhi + r) * OUT_DIM + n]);
        }
    }
}

extern "C" void kernel_launch(void* const* d_in, const int* in_sizes, int n_in,
                              void* d_out, int out_size, void* d_ws, size_t ws_size,
                              hipStream_t stream) {
    const float* x     = (const float*)d_in[0];
    const float* shift = (const float*)d_in[1];
    const float* sign  = (const float*)d_in[2];
    const float* bias  = (const float*)d_in[3];
    float* out = (float*)d_out;
    _Float16* wp = (_Float16*)d_ws;    // 131072 bytes of scratch used

    // 1) Pack W into WMMA B-fragment layout (4096 threads, trivial cost).
    pack_w_kernel<<<(KT * NT * 32 + 255) / 256, 256, 0, stream>>>(shift, sign, wp);

    // 2) Memory-bound WMMA GEMM over the batch; W staged through LDS.
    const int batch  = in_sizes[0] / IN_DIM;          // 524288
    const int blocks = batch / ROWS_PER_BLOCK;        // 4096
    denseshift_gemm<<<blocks, WAVES * 32, 0, stream>>>(x, wp, bias, out);
}
